// Decoder_31387620999286
// MI455X (gfx1250) — matrix operand order repaired and checked by reading, not compiled, not run on hardware
//
#include <hip/hip_runtime.h>
#include <hip/hip_bf16.h>
#include <stdint.h>

// ---------------------------------------------------------------------------
// GPT decoder forward for MI455X (gfx1250):
//   - all GEMM operands pre-staged as bf16 in global memory
//   - GEMM = async DMA (global_load_async_to_lds_b128) -> v_wmma_f32_16x16x32_bf16
//   - fp32 kept for residual stream / attn maps / logits / reductions
// ---------------------------------------------------------------------------

typedef __attribute__((ext_vector_type(16))) __bf16 v16bf;
typedef __attribute__((ext_vector_type(8)))  float  v8f;

namespace {
constexpr int D_   = 512;
constexpr int T_   = 1024;
constexpr int B_   = 2;
constexpr int H_   = 8;
constexpr int HS_  = 64;
constexpr int V_   = 32000;
constexpr int L_   = 4;
constexpr int FF_  = 2048;
constexpr int M_   = B_ * T_;          // 2048 token rows
}

// fp32 -> bf16 bits (native cvt)
__device__ __forceinline__ unsigned short f2bfu(float f) {
  union { __bf16 h; unsigned short s; } v;
  v.h = (__bf16)f;
  return v.s;
}

// ---------------------------------------------------------------------------
// Async-pipelined GEMM:
//   C(fp32, optional) = alpha * A x B^T + bias + resid  (relu optional)
//   CB(bf16, optional) = same value (optionally transposed store)
// A: bf16 [M][K] row-major (lda).  Bt: bf16 [N][K] row-major (ldb).
// Block tile 64x64, K-step 64, double-buffered LDS fed by
// global_load_async_to_lds_b128 (ASYNCcnt), 8 waves x 4 WMMA per k-iter.
// gridDim = (N/64, M/64, nBatch); batch advances A/Bt/C/CB by element strides.
// ---------------------------------------------------------------------------
#define TILE_M 64
#define TILE_N 64
#define TILE_K 64
#define LDS_STR 72   // elements: rows 144B (16B-aligned for b128, 36-dword stride)

__global__ __launch_bounds__(256) void gemm_async_bf16_wmma(
    const unsigned short* __restrict__ A, int lda, long long aBS,
    const unsigned short* __restrict__ Bt, int ldb, long long bBS,
    const float* __restrict__ bias,
    const float* __restrict__ resid, int ldr,
    float* __restrict__ C, int ldc, long long cBS,
    unsigned short* __restrict__ CB, int ldcb, long long cbBS, int cbTrans,
    int Kdim, float alpha, int relu)
{
  __shared__ unsigned short sA[2][TILE_M * LDS_STR];
  __shared__ unsigned short sB[2][TILE_N * LDS_STR];

  A  += (size_t)blockIdx.z * (size_t)aBS;
  Bt += (size_t)blockIdx.z * (size_t)bBS;
  if (C)  C  += (size_t)blockIdx.z * (size_t)cBS;
  if (CB) CB += (size_t)blockIdx.z * (size_t)cbBS;

  const int tid  = threadIdx.x;
  const int lane = tid & 31;
  const int wave = tid >> 5;
  const int wm   = wave >> 1;      // 0..3 : 16-row slice inside 64
  const int wn   = wave & 1;       // 0..1 : 32-col slice inside 64
  const int row0 = blockIdx.y * TILE_M;
  const int col0 = blockIdx.x * TILE_N;

  const int half = lane >> 4;
  const int l15  = lane & 15;
  const int mrow  = 16 * wm + l15;
  const int nrow0 = 32 * wn + l15;
  const int nrow1 = 32 * wn + 16 + l15;

  v8f acc0 = {};
  v8f acc1 = {};

  // async-stage a 64x64 bf16 tile of A and of B^T into LDS buffer `buf`
  // (2 x b128 per thread per operand; ASYNCcnt-tracked DMA, no VGPR data)
  auto stage = [&](int buf, int k0) {
#pragma unroll
    for (int j = 0; j < 2; ++j) {
      int idx = j * 256 + tid;
      int r = idx >> 3, c8 = idx & 7;           // row, 8-elem (16B) chunk
      unsigned lA = (unsigned)(size_t)(const void*)&sA[buf][r * LDS_STR + c8 * 8];
      const unsigned short* gA = A + (size_t)(row0 + r) * lda + k0 + c8 * 8;
      asm volatile("global_load_async_to_lds_b128 %0, %1, off"
                   :: "v"(lA), "v"(gA) : "memory");
      unsigned lB = (unsigned)(size_t)(const void*)&sB[buf][r * LDS_STR + c8 * 8];
      const unsigned short* gB = Bt + (size_t)(col0 + r) * ldb + k0 + c8 * 8;
      asm volatile("global_load_async_to_lds_b128 %0, %1, off"
                   :: "v"(lB), "v"(gB) : "memory");
    }
  };

  const int nk = Kdim / TILE_K;
  stage(0, 0);
  int cur = 0;

  for (int it = 0; it < nk; ++it) {
    asm volatile("s_wait_asynccnt 0" ::: "memory");   // this wave's DMAs done
    __syncthreads();                                  // all waves' DMAs done
    if (it + 1 < nk) stage(cur ^ 1, (it + 1) * TILE_K);

    const unsigned short* a_ = &sA[cur][mrow * LDS_STR];
    const unsigned short* b0 = &sB[cur][nrow0 * LDS_STR];
    const unsigned short* b1 = &sB[cur][nrow1 * LDS_STR];

#pragma unroll
    for (int kk = 0; kk < TILE_K; kk += 32) {
      v16bf af, bf0, bf1;
      union { unsigned u; __bf16 h[2]; } cv;
#pragma unroll
      for (int v = 0; v < 8; ++v) {
        // A 16x32 layout: vgpr 0..3 -> K=8*half+2v.. ; vgpr 4..7 -> K=16+8*half+..
        int ka = kk + ((v < 4) ? 0 : 16) + 8 * half + 2 * (v & 3);
        cv.u = *(const unsigned*)&a_[ka];
        af[2 * v] = cv.h[0]; af[2 * v + 1] = cv.h[1];
        // B (B^T rows): same K mapping as A
        int kb = kk + ((v < 4) ? 0 : 16) + 8 * half + 2 * (v & 3);
        cv.u = *(const unsigned*)&b0[kb];
        bf0[2 * v] = cv.h[0]; bf0[2 * v + 1] = cv.h[1];
        cv.u = *(const unsigned*)&b1[kb];
        bf1[2 * v] = cv.h[0]; bf1[2 * v + 1] = cv.h[1];
      }
      acc0 = __builtin_amdgcn_wmma_f32_16x16x32_bf16(false, af, false, bf0,
                                                     (short)0, acc0, false, false);
      acc1 = __builtin_amdgcn_wmma_f32_16x16x32_bf16(false, af, false, bf1,
                                                     (short)0, acc1, false, false);
    }
    cur ^= 1;
  }

  // epilogue: C/D layout: vgpr r -> M = r + 8*half, N = lane&15
  const int n0 = col0 + 32 * wn + l15;
  const int n1 = n0 + 16;
  const float bias0 = bias ? bias[n0] : 0.0f;
  const float bias1 = bias ? bias[n1] : 0.0f;
#pragma unroll
  for (int r = 0; r < 8; ++r) {
    int m = row0 + 16 * wm + r + 8 * half;
    float v0 = alpha * acc0[r] + bias0;
    float v1 = alpha * acc1[r] + bias1;
    if (resid) {
      v0 += resid[(size_t)m * ldr + n0];
      v1 += resid[(size_t)m * ldr + n1];
    }
    if (relu) { v0 = fmaxf(v0, 0.0f); v1 = fmaxf(v1, 0.0f); }
    if (C) {
      C[(size_t)m * ldc + n0] = v0;
      C[(size_t)m * ldc + n1] = v1;
    }
    if (CB) {
      if (!cbTrans) {
        CB[(size_t)m * ldcb + n0] = f2bfu(v0);
        CB[(size_t)m * ldcb + n1] = f2bfu(v1);
      } else {
        CB[(size_t)n0 * ldcb + m] = f2bfu(v0);
        CB[(size_t)n1 * ldcb + m] = f2bfu(v1);
      }
    }
  }
}

// ---------------------------------------------------------------------------
// Batched transpose+convert: in fp32 [Kd][Nd] -> out bf16 [Nd][Kd]
// grid (Nd/32, Kd/32, batch)
// ---------------------------------------------------------------------------
__global__ __launch_bounds__(256) void convT_kernel(
    const float* __restrict__ in, int Kd, int Nd, long long inBS,
    unsigned short* __restrict__ out, long long outBS)
{
  __shared__ float t[32][33];
  in  += (size_t)blockIdx.z * (size_t)inBS;
  out += (size_t)blockIdx.z * (size_t)outBS;
  int n0 = blockIdx.x * 32, k0 = blockIdx.y * 32;
  int c = threadIdx.x & 31, r = threadIdx.x >> 5;
#pragma unroll
  for (int j = 0; j < 4; ++j)
    t[r + 8 * j][c] = in[(size_t)(k0 + r + 8 * j) * Nd + (n0 + c)];
  __syncthreads();
#pragma unroll
  for (int j = 0; j < 4; ++j)
    out[(size_t)(n0 + r + 8 * j) * Kd + (k0 + c)] = f2bfu(t[c][r + 8 * j]);
}

// ---------------------------------------------------------------------------
// x[row] = tok_emb[idx[row]] + pos_emb[row % T]   (fp32 residual stream)
// ---------------------------------------------------------------------------
__global__ __launch_bounds__(256) void embed_kernel(
    const int* __restrict__ idx, const float* __restrict__ tok,
    const float* __restrict__ pos, float* __restrict__ x)
{
  int row = blockIdx.x;
  int t   = row & (T_ - 1);
  int token = idx[row];
  int d = threadIdx.x;
  const float* tr = tok + (size_t)token * D_;
  const float* pr = pos + (size_t)t * D_;
  float* xr = x + (size_t)row * D_;
  xr[d]       = tr[d]       + pr[d];
  xr[d + 256] = tr[d + 256] + pr[d + 256];
}

// ---------------------------------------------------------------------------
// LayerNorm over D=512 per row; emits bf16 (GEMM A operand)
// ---------------------------------------------------------------------------
__global__ __launch_bounds__(256) void ln_kernel(
    const float* __restrict__ x, const float* __restrict__ g,
    const float* __restrict__ b, unsigned short* __restrict__ ybf)
{
  __shared__ float red[256];
  int row = blockIdx.x;
  int tid = threadIdx.x;
  const float* xr = x + (size_t)row * D_;
  float v0 = xr[tid], v1 = xr[tid + 256];

  red[tid] = v0 + v1;
  __syncthreads();
  for (int s = 128; s > 0; s >>= 1) {
    if (tid < s) red[tid] += red[tid + s];
    __syncthreads();
  }
  float mean = red[0] * (1.0f / D_);
  __syncthreads();

  float d0 = v0 - mean, d1 = v1 - mean;
  red[tid] = d0 * d0 + d1 * d1;
  __syncthreads();
  for (int s = 128; s > 0; s >>= 1) {
    if (tid < s) red[tid] += red[tid + s];
    __syncthreads();
  }
  float inv = rsqrtf(red[0] * (1.0f / D_) + 1e-5f);

  unsigned short* yr = ybf + (size_t)row * D_;
  yr[tid]       = f2bfu(d0 * inv * g[tid]       + b[tid]);
  yr[tid + 256] = f2bfu(d1 * inv * g[tid + 256] + b[tid + 256]);
}

// ---------------------------------------------------------------------------
// In-place causal softmax on fp32 scores (final attn maps) + bf16 copy
// ---------------------------------------------------------------------------
__global__ __launch_bounds__(256) void attn_softmax_kernel(
    float* __restrict__ attn_l, unsigned short* __restrict__ wbf)
{
  __shared__ float red[256];
  int r   = blockIdx.x;            // (h*B + b)*T + t
  int t   = r & (T_ - 1);
  int tid = threadIdx.x;
  float* row = attn_l + (size_t)r * T_;
  unsigned short* wrow = wbf + (size_t)r * T_;

  float vals[4];
  float mx = -1e30f;
#pragma unroll
  for (int j = 0; j < 4; ++j) {
    int col = j * 256 + tid;
    float v = (col <= t) ? row[col] : -1e30f;
    vals[j] = v;
    mx = fmaxf(mx, v);
  }
  red[tid] = mx;
  __syncthreads();
  for (int s = 128; s > 0; s >>= 1) {
    if (tid < s) red[tid] = fmaxf(red[tid], red[tid + s]);
    __syncthreads();
  }
  mx = red[0];
  __syncthreads();

  float sum = 0.0f;
#pragma unroll
  for (int j = 0; j < 4; ++j) {
    int col = j * 256 + tid;
    float e = (col <= t) ? __expf(vals[j] - mx) : 0.0f;
    vals[j] = e;
    sum += e;
  }
  red[tid] = sum;
  __syncthreads();
  for (int s = 128; s > 0; s >>= 1) {
    if (tid < s) red[tid] += red[tid + s];
    __syncthreads();
  }
  float inv = 1.0f / red[0];
#pragma unroll
  for (int j = 0; j < 4; ++j) {
    int col = j * 256 + tid;
    float w = vals[j] * inv;
    row[col]  = w;
    wrow[col] = f2bfu(w);
  }
}

// ---------------------------------------------------------------------------
// Per-row: rowlp[row] = logits[row, target] - logsumexp(logits[row,:])
// ---------------------------------------------------------------------------
__global__ __launch_bounds__(256) void row_logprob_kernel(
    const float* __restrict__ logits, const int* __restrict__ targets,
    float* __restrict__ rowlp)
{
  __shared__ float red[256];
  int row = blockIdx.x;
  int tid = threadIdx.x;
  const float* lr = logits + (size_t)row * V_;

  float mx = -1e30f;
  for (int c = tid; c < V_; c += 256) mx = fmaxf(mx, lr[c]);
  red[tid] = mx;
  __syncthreads();
  for (int s = 128; s > 0; s >>= 1) {
    if (tid < s) red[tid] = fmaxf(red[tid], red[tid + s]);
    __syncthreads();
  }
  mx = red[0];
  __syncthreads();

  float sum = 0.0f;
  for (int c = tid; c < V_; c += 256) sum += expf(lr[c] - mx);
  red[tid] = sum;
  __syncthreads();
  for (int s = 128; s > 0; s >>= 1) {
    if (tid < s) red[tid] += red[tid + s];
    __syncthreads();
  }
  if (tid == 0) {
    float logZ = mx + logf(red[0]);
    rowlp[row] = lr[targets[row]] - logZ;
  }
}

__global__ __launch_bounds__(256) void loss_reduce_kernel(
    const float* __restrict__ rowlp, float* __restrict__ loss)
{
  __shared__ float red[256];
  int tid = threadIdx.x;
  float s = 0.0f;
#pragma unroll
  for (int j = 0; j < M_ / 256; ++j) s += rowlp[j * 256 + tid];
  red[tid] = s;
  __syncthreads();
  for (int st = 128; st > 0; st >>= 1) {
    if (tid < st) red[tid] += red[tid + st];
    __syncthreads();
  }
  if (tid == 0) loss[0] = -red[0] / (float)M_;
}

// ---------------------------------------------------------------------------
// Host-side orchestration
// ---------------------------------------------------------------------------
extern "C" void kernel_launch(void* const* d_in, const int* in_sizes, int n_in,
                              void* d_out, int out_size, void* d_ws, size_t ws_size,
                              hipStream_t stream) {
  (void)in_sizes; (void)n_in; (void)out_size; (void)ws_size;

  const int*   idx     = (const int*)  d_in[0];
  const int*   targets = (const int*)  d_in[1];
  const float* tok_emb = (const float*)d_in[2];
  const float* pos_emb = (const float*)d_in[3];
  const float* wq      = (const float*)d_in[4];
  const float* wk      = (const float*)d_in[5];
  const float* wv      = (const float*)d_in[6];
  const float* wproj   = (const float*)d_in[7];
  const float* bproj   = (const float*)d_in[8];
  const float* ln1_g   = (const float*)d_in[9];
  const float* ln1_b   = (const float*)d_in[10];
  const float* ln2_g   = (const float*)d_in[11];
  const float* ln2_b   = (const float*)d_in[12];
  const float* w1      = (const float*)d_in[13];
  const float* b1      = (const float*)d_in[14];
  const float* w2      = (const float*)d_in[15];
  const float* b2      = (const float*)d_in[16];
  const float* lnf_g   = (const float*)d_in[17];
  const float* lnf_b   = (const float*)d_in[18];
  const float* lm_w    = (const float*)d_in[19];
  const float* lm_b    = (const float*)d_in[20];

  float* out    = (float*)d_out;
  float* logits = out;                                        // [2,1024,32000]
  float* loss   = out + (size_t)M_ * V_;                      // scalar
  float* attn   = loss + 1;                                   // [L,H,B,T,T]

  // ---- workspace carve ----
  float* wsf   = (float*)d_ws;
  float* x     = wsf;                       // [2048,512] fp32 residual stream
  float* rowlp = x + (size_t)M_ * D_;       // [2048]
  unsigned short* us = (unsigned short*)(rowlp + 2048);
  unsigned short* xnbf   = us;  us += (size_t)M_ * D_;        // [2048,512]
  unsigned short* qbf    = us;  us += (size_t)M_ * D_;        // [H][2048][64]
  unsigned short* kbf    = us;  us += (size_t)M_ * D_;        // [H][2048][64]
  unsigned short* vtbf   = us;  us += (size_t)M_ * D_;        // [H][64][2048]
  unsigned short* outbbf = us;  us += (size_t)M_ * D_;        // [2048,512]
  unsigned short* hbf    = us;  us += (size_t)M_ * FF_;       // [2048,2048]
  unsigned short* weibf  = us;  us += (size_t)H_ * B_ * T_ * T_; // bf16 attn copy
  unsigned short* wqT    = us;  us += (size_t)L_ * H_ * HS_ * D_; // [L*H][64][512]
  unsigned short* wkT    = us;  us += (size_t)L_ * H_ * HS_ * D_;
  unsigned short* wvT    = us;  us += (size_t)L_ * H_ * HS_ * D_;
  unsigned short* wprojT = us;  us += (size_t)L_ * D_ * D_;      // [L][512][512]
  unsigned short* w1T    = us;  us += (size_t)L_ * FF_ * D_;     // [L][2048][512]
  unsigned short* w2T    = us;  us += (size_t)L_ * D_ * FF_;     // [L][512][2048]
  unsigned short* lmwT   = us;  us += (size_t)V_ * D_;           // [32000][512]

  const long long perHeadQ = (long long)M_ * HS_;     // 131072
  const long long perBHQ   = (long long)T_ * HS_;     // 65536
  const long long perBHA   = (long long)T_ * T_;      // 1048576

  // ---- one-time (per call) weight transpose+convert to bf16 [N][K] ----
  convT_kernel<<<dim3(HS_ / 32, D_ / 32, L_ * H_), 256, 0, stream>>>(
      wq, D_, HS_, (long long)D_ * HS_, wqT, (long long)HS_ * D_);
  convT_kernel<<<dim3(HS_ / 32, D_ / 32, L_ * H_), 256, 0, stream>>>(
      wk, D_, HS_, (long long)D_ * HS_, wkT, (long long)HS_ * D_);
  convT_kernel<<<dim3(HS_ / 32, D_ / 32, L_ * H_), 256, 0, stream>>>(
      wv, D_, HS_, (long long)D_ * HS_, wvT, (long long)HS_ * D_);
  convT_kernel<<<dim3(D_ / 32, D_ / 32, L_), 256, 0, stream>>>(
      wproj, D_, D_, (long long)D_ * D_, wprojT, (long long)D_ * D_);
  convT_kernel<<<dim3(FF_ / 32, D_ / 32, L_), 256, 0, stream>>>(
      w1, D_, FF_, (long long)D_ * FF_, w1T, (long long)FF_ * D_);
  convT_kernel<<<dim3(D_ / 32, FF_ / 32, L_), 256, 0, stream>>>(
      w2, FF_, D_, (long long)FF_ * D_, w2T, (long long)D_ * FF_);
  convT_kernel<<<dim3(V_ / 32, D_ / 32, 1), 256, 0, stream>>>(
      lm_w, D_, V_, 0LL, lmwT, 0LL);

  // ---- embeddings ----
  embed_kernel<<<M_, 256, 0, stream>>>(idx, tok_emb, pos_emb, x);

  for (int l = 0; l < L_; ++l) {
    float* attn_l = attn + (size_t)l * H_ * B_ * T_ * T_;

    // ln1 -> bf16
    ln_kernel<<<M_, 256, 0, stream>>>(x, ln1_g + l * D_, ln1_b + l * D_, xnbf);

    // Q/K/V projections (batched over heads); q,k bf16; v stored transposed bf16
    {
      dim3 g(HS_ / TILE_N, M_ / TILE_M, H_);
      const unsigned short* wqTl = wqT + (size_t)l * H_ * HS_ * D_;
      const unsigned short* wkTl = wkT + (size_t)l * H_ * HS_ * D_;
      const unsigned short* wvTl = wvT + (size_t)l * H_ * HS_ * D_;
      gemm_async_bf16_wmma<<<g, 256, 0, stream>>>(
          xnbf, D_, 0LL, wqTl, D_, (long long)HS_ * D_,
          nullptr, nullptr, 0, nullptr, 0, 0LL,
          qbf, HS_, perHeadQ, /*cbTrans=*/0, D_, 1.0f, 0);
      gemm_async_bf16_wmma<<<g, 256, 0, stream>>>(
          xnbf, D_, 0LL, wkTl, D_, (long long)HS_ * D_,
          nullptr, nullptr, 0, nullptr, 0, 0LL,
          kbf, HS_, perHeadQ, /*cbTrans=*/0, D_, 1.0f, 0);
      gemm_async_bf16_wmma<<<g, 256, 0, stream>>>(
          xnbf, D_, 0LL, wvTl, D_, (long long)HS_ * D_,
          nullptr, nullptr, 0, nullptr, 0, 0LL,
          vtbf, M_, perHeadQ, /*cbTrans=*/1, D_, 1.0f, 0);
    }

    // scores = Q K^T / 8, straight into the attn_maps slot (fp32)
    {
      dim3 g(T_ / TILE_N, T_ / TILE_M, H_ * B_);
      gemm_async_bf16_wmma<<<g, 256, 0, stream>>>(
          qbf, HS_, perBHQ, kbf, HS_, perBHQ,
          nullptr, nullptr, 0,
          attn_l, T_, perBHA,
          nullptr, 0, 0LL, 0, HS_, 0.125f, 0);
    }

    // causal softmax in place (fp32 maps) + bf16 copy for next GEMM
    attn_softmax_kernel<<<H_ * B_ * T_, 256, 0, stream>>>(attn_l, weibf);

    // out = wei @ V  -> bf16 heads-concat outb; per-head, batched over b
    for (int h = 0; h < H_; ++h) {
      dim3 g(HS_ / TILE_N, T_ / TILE_M, B_);
      const unsigned short* Aw = weibf + (size_t)h * B_ * perBHA;
      const unsigned short* Bv = vtbf + (size_t)h * perHeadQ;   // [64][2048]
      unsigned short* Co = outbbf + (size_t)h * HS_;
      gemm_async_bf16_wmma<<<g, 256, 0, stream>>>(
          Aw, T_, perBHA, Bv, M_, (long long)T_,
          nullptr, nullptr, 0, nullptr, 0, 0LL,
          Co, D_, (long long)T_ * D_, /*cbTrans=*/0, T_, 1.0f, 0);
    }

    // x = x + outb @ wproj + bproj   (fp32 residual update)
    {
      dim3 g(D_ / TILE_N, M_ / TILE_M, 1);
      gemm_async_bf16_wmma<<<g, 256, 0, stream>>>(
          outbbf, D_, 0LL, wprojT + (size_t)l * D_ * D_, D_, 0LL,
          bproj + l * D_, x, D_, x, D_, 0LL,
          nullptr, 0, 0LL, 0, D_, 1.0f, 0);
    }

    // ln2 -> bf16
    ln_kernel<<<M_, 256, 0, stream>>>(x, ln2_g + l * D_, ln2_b + l * D_, xnbf);

    // h = relu(xn @ w1 + b1) -> bf16 only
    {
      dim3 g(FF_ / TILE_N, M_ / TILE_M, 1);
      gemm_async_bf16_wmma<<<g, 256, 0, stream>>>(
          xnbf, D_, 0LL, w1T + (size_t)l * FF_ * D_, D_, 0LL,
          b1 + (size_t)l * FF_, nullptr, 0, nullptr, 0, 0LL,
          hbf, FF_, 0LL, 0, D_, 1.0f, /*relu=*/1);
    }

    // x = x + h @ w2 + b2
    {
      dim3 g(D_ / TILE_N, M_ / TILE_M, 1);
      gemm_async_bf16_wmma<<<g, 256, 0, stream>>>(
          hbf, FF_, 0LL, w2T + (size_t)l * D_ * FF_, FF_, 0LL,
          b2 + (size_t)l * D_, x, D_, x, D_, 0LL,
          nullptr, 0, 0LL, 0, FF_, 1.0f, 0);
    }
  }

  // final LN -> bf16
  ln_kernel<<<M_, 256, 0, stream>>>(x, lnf_g, lnf_b, xnbf);

  // logits = xn @ lm_w + lm_b   (fp32 to d_out)
  {
    dim3 g(V_ / TILE_N, M_ / TILE_M, 1);
    gemm_async_bf16_wmma<<<g, 256, 0, stream>>>(
        xnbf, D_, 0LL, lmwT, D_, 0LL,
        lm_b, nullptr, 0, logits, V_, 0LL,
        nullptr, 0, 0LL, 0, D_, 1.0f, 0);
  }

  // loss = -mean(logp[target])
  row_logprob_kernel<<<M_, 256, 0, stream>>>(logits, targets, rowlp);
  loss_reduce_kernel<<<1, 256, 0, stream>>>(rowlp, loss);
}
